// Net_66451734004145
// MI455X (gfx1250) — matrix operand order repaired and checked by reading, not compiled, not run on hardware
//
#include <hip/hip_runtime.h>
#include <hip/hip_bf16.h>
#include <math.h>

#define SEQ 128
#define BAT 16
#define INP 64
#define HG  100
#define G3  300       // 3*H
#define HID 256
#define NOUT 2
#define NPAIR (SEQ*SEQ)        // 16384
#define NROWS (NPAIR*BAT)      // 262144

typedef _Float16 v8h  __attribute__((ext_vector_type(8)));
typedef _Float16 v16h __attribute__((ext_vector_type(16)));
typedef float    v8f  __attribute__((ext_vector_type(8)));

__device__ __forceinline__ float sigmoidf_(float x){ return 1.0f/(1.0f+expf(-x)); }

// ---------------------------------------------------------------------------
// Kernel 0: input gates gi = x @ W_ih^T + b_ih for ALL timesteps (h-independent,
// fully parallel) -- pulled off the GRU's serial critical path.
// ---------------------------------------------------------------------------
__global__ void gia_kernel(const float* __restrict__ x, const float* __restrict__ Wih,
                           const float* __restrict__ bih, float* __restrict__ gia) {
    const int row = blockIdx.x;          // s*BAT + b, 2048 rows
    const int t   = threadIdx.x;         // 320 threads, 300 active
    if (t < G3) {
        const float* xr = &x[row*INP];
        const float* wi = &Wih[t*INP];
        float a = bih[t];
        #pragma unroll 8
        for (int c = 0; c < INP; ++c) a = fmaf(xr[c], wi[c], a);
        gia[row*G3 + t] = a;
    }
}

// ---------------------------------------------------------------------------
// Kernel 1: GRU recurrence. One workgroup per batch element (the only parallel
// axis). Per step only the h-dependent half (100 MACs/gate) remains serial.
// ---------------------------------------------------------------------------
__global__ void gru_kernel(const float* __restrict__ gia, const float* __restrict__ h0,
                           const float* __restrict__ Whh, const float* __restrict__ bhh,
                           float* __restrict__ out) {
    const int b = blockIdx.x;
    const int t = threadIdx.x;
    __shared__ float h[HG];
    __shared__ float gi[G3];
    __shared__ float gh[G3];
    if (t < HG) h[t] = h0[b*HG + t];
    __syncthreads();
    for (int s = 0; s < SEQ; ++s) {
        if (t < G3) {
            const float* wh = &Whh[t*HG];
            float g = bhh[t];
            #pragma unroll 4
            for (int c = 0; c < HG; ++c) g = fmaf(h[c], wh[c], g);
            gi[t] = gia[(s*BAT + b)*G3 + t];
            gh[t] = g;
        }
        __syncthreads();
        if (t < HG) {
            float r  = sigmoidf_(gi[t]      + gh[t]);
            float z  = sigmoidf_(gi[HG+t]   + gh[HG+t]);
            float n  = tanhf(gi[2*HG+t] + r*gh[2*HG+t]);
            float hn = (1.0f - z)*n + z*h[t];
            h[t] = hn;
            out[(s*BAT + b)*HG + t] = hn;
        }
        __syncthreads();
    }
}

// ---------------------------------------------------------------------------
// Kernel 2: P = out @ W1[:, :100]^T ; Q = out @ W1[:, 100:]^T   (2048x256, K=100)
// Collapses the 26.8 GFLOP layer-1 GEMM into 0.2 GFLOP via the adjacency identity.
// ---------------------------------------------------------------------------
__global__ void pq_kernel(const float* __restrict__ gout, const float* __restrict__ W1,
                          float* __restrict__ P, float* __restrict__ Q) {
    const int row = blockIdx.x;      // s*BAT + b, 2048 rows
    const int n   = threadIdx.x;     // 256 cols
    const float* orow = &gout[row*HG];
    const float* w    = &W1[n*(2*HG)];
    float p = 0.f, q = 0.f;
    #pragma unroll 4
    for (int k = 0; k < HG; ++k) { float o = orow[k]; p = fmaf(o, w[k], p); q = fmaf(o, w[HG+k], q); }
    P[row*HID + n] = p;
    Q[row*HID + n] = q;
}

// ---------------------------------------------------------------------------
// Kernel 3a: W2 (256x256 f32, n-major) -> f16 k-major (transposed): a WMMA
// B-fragment is 16 f16 LDS reads per lane at stride HID (lane%16 = N col).
// ---------------------------------------------------------------------------
__global__ void w2t_kernel(const float* __restrict__ W2, _Float16* __restrict__ w2t) {
    int idx = blockIdx.x*256 + threadIdx.x;      // 65536
    int k = idx >> 8, n = idx & 255;
    w2t[idx] = (_Float16)W2[n*HID + k];
}

// Kernel 3b: W3 (10x256 f32) -> f16 k-major 256x16 tile, cols 10..15 zero-padded,
// so layer 3 runs on WMMA too.
__global__ void w3h_kernel(const float* __restrict__ W3, _Float16* __restrict__ w3h) {
    int idx = blockIdx.x*256 + threadIdx.x;      // 4096
    int k = idx >> 4, n = idx & 15;
    w3h[idx] = (n < 10) ? (_Float16)W3[n*HID + k] : (_Float16)0.f;
}

// ---------------------------------------------------------------------------
// Kernel 4: fused h1-construct -> [256x256 WMMA] -> relu -> [256x16 WMMA] -> head.
// Block = 256 threads (8 waves), 128 rows = 8 (i,j) pairs x 16 batches; i is
// block-constant, so the 16 P[i,b] rows are staged once in LDS (b1 pre-added).
// LDS: h1 64K + W2^T 128K + h2 64K + W3 8K + Pstage 16K + h3 5K = 285 KB.
// Layer-2 N dim in TWO passes of 8 tiles (acc[8] = 64 VGPRs live): no spills.
// ---------------------------------------------------------------------------
__global__ void __launch_bounds__(256) mlp_kernel(
        const float* __restrict__ P, const float* __restrict__ Q,
        const _Float16* __restrict__ w2t, const _Float16* __restrict__ w3h,
        const float* __restrict__ b1, const float* __restrict__ b2,
        const float* __restrict__ b3,
        const float* __restrict__ Wt, const float* __restrict__ bt,
        float* __restrict__ logits) {
    __shared__ _Float16 h1s[128*HID];    //  64 KB: layer-1 activations (f16)
    __shared__ _Float16 w2s[HID*HID];    // 128 KB: W2^T, k-major
    __shared__ _Float16 h2s[128*HID];    //  64 KB: layer-2 activations (f16)
    __shared__ _Float16 w3s[HID*16];     //   8 KB: W3^T, k-major, padded to 16
    __shared__ float    ps[BAT*HID];     //  16 KB: P[i,b,:] + b1 (i block-constant)
    __shared__ float    h3s[8*16*10];    //   5 KB: per-wave layer-3 activations

    const int tid  = threadIdx.x;
    const int lane = tid & 31;
    const int wave = tid >> 5;
    const int R0   = blockIdx.x * 128;       // first global row of this block
    const int iblk = blockIdx.x >> 4;        // i index (same for all 128 rows)

    // Stage W2^T / W3^T (16B vector copies) and P rows (+b1) into LDS.
    {
        const v8h* src = (const v8h*)w2t;
        v8h*       dst = (v8h*)w2s;
        for (int e = tid; e < (HID*HID)/8; e += 256) dst[e] = src[e];
        const v8h* s3 = (const v8h*)w3h;
        v8h*       d3 = (v8h*)w3s;
        for (int e = tid; e < (HID*16)/8; e += 256) d3[e] = s3[e];
        for (int e = tid; e < BAT*HID; e += 256) {
            int b = e >> 8, n = e & 255;
            ps[e] = P[(iblk*BAT + b)*HID + n] + b1[n];
        }
    }
    __syncthreads();
    // Build h1 = relu(Pstage[b] + Q[j,b]) for 128 rows; thread owns column tid.
    for (int r = 0; r < 128; ++r) {
        int Rg = R0 + r;
        int p  = Rg >> 4;  int b = Rg & 15;      // row = (i*S + j)*B + b
        int j  = p  & 127;
        float v = ps[b*HID + tid] + Q[(j*BAT + b)*HID + tid];
        h1s[r*HID + tid] = (_Float16)fmaxf(v, 0.f);
    }
    __syncthreads();

    const int arow   = wave*16 + (lane & 15);
    const int koffA0 = (lane < 16) ? 0 : 8;      // 16-bit A layout: lane halves
    const int koffA1 = koffA0 + 16;
    const int mb     = (lane < 16) ? 0 : 8;      // C layout: M base per lane half
    const int ncol   = lane & 15;                // C layout: N = lane%16

    // ---- Layer 2: 16x256 strip per wave, two N-halves of 8 tiles each.
    #pragma unroll 1
    for (int half = 0; half < 2; ++half) {
        v8f acc[8];
        #pragma unroll
        for (int nt = 0; nt < 8; ++nt) { v8f z = {}; acc[nt] = z; }
        #pragma unroll 1
        for (int kk = 0; kk < 8; ++kk) {         // K = 256 in chunks of 32
            union { v16h v; v8h h[2]; } ua;
            ua.h[0] = *(const v8h*)&h1s[arow*HID + kk*32 + koffA0];
            ua.h[1] = *(const v8h*)&h1s[arow*HID + kk*32 + koffA1];
            const _Float16* bbase = &w2s[(kk*32)*HID + half*128 + (lane & 15)];
            #pragma unroll
            for (int nt = 0; nt < 8; ++nt) {
                v16h bf;
                #pragma unroll
                for (int s = 0; s < 16; ++s)
                    bf[s] = bbase[((s < 8) ? (koffA0 + s) : (koffA1 + s - 8))*HID + nt*16];
                acc[nt] = __builtin_amdgcn_wmma_f32_16x16x32_f16(
                    false, ua.v, false, bf, (short)0, acc[nt], false, false);
            }
        }
        // h2 = relu(acc + b2) for this N-half.
        #pragma unroll
        for (int nt = 0; nt < 8; ++nt) {
            int   col  = half*128 + nt*16 + ncol;
            float bias = b2[col];
            #pragma unroll
            for (int v = 0; v < 8; ++v)
                h2s[(wave*16 + mb + v)*HID + col] = (_Float16)fmaxf(acc[nt][v] + bias, 0.f);
        }
    }
    __syncthreads();

    // ---- Layer 3 on WMMA: 16x16 output tile per wave (cols 10..15 are garbage).
    {
        v8f acc3 = {};
        #pragma unroll 1
        for (int kk = 0; kk < 8; ++kk) {
            union { v16h v; v8h h[2]; } ua;
            ua.h[0] = *(const v8h*)&h2s[arow*HID + kk*32 + koffA0];
            ua.h[1] = *(const v8h*)&h2s[arow*HID + kk*32 + koffA1];
            v16h bf;
            #pragma unroll
            for (int s = 0; s < 16; ++s)
                bf[s] = w3s[(kk*32 + ((s < 8) ? (koffA0 + s) : (koffA1 + s - 8)))*16 + ncol];
            acc3 = __builtin_amdgcn_wmma_f32_16x16x32_f16(
                false, ua.v, false, bf, (short)0, acc3, false, false);
        }
        if (ncol < 10) {
            float bias = b3[ncol];
            #pragma unroll
            for (int v = 0; v < 8; ++v)
                h3s[wave*160 + (mb + v)*10 + ncol] = fmaxf(acc3[v] + bias, 0.f);
        }
    }
    __syncthreads();

    // ---- Head: 10 -> 2 per row; 128 rows x 2 = 256 = blockDim.
    {
        int rl = tid >> 1, o = tid & 1;
        int w  = rl >> 4,  r = rl & 15;
        const float* h3 = &h3s[w*160 + r*10];
        float a = bt[o];
        #pragma unroll
        for (int u = 0; u < 10; ++u) a = fmaf(h3[u], Wt[o*10 + u], a);
        logits[(R0 + rl)*NOUT + o] = a;
    }
}

// ---------------------------------------------------------------------------
// Kernel 5: log_softmax over axis 0 (16384) for each of the 32 (b,out) columns.
// ---------------------------------------------------------------------------
__global__ void lsm_kernel(const float* __restrict__ logits, float* __restrict__ out) {
    const int c   = blockIdx.x;      // b*2 + o, 32 columns
    const int tid = threadIdx.x;
    __shared__ float red[256];
    float m = -INFINITY;
    for (int p = tid; p < NPAIR; p += 256) m = fmaxf(m, logits[p*32 + c]);
    red[tid] = m; __syncthreads();
    for (int s = 128; s > 0; s >>= 1) { if (tid < s) red[tid] = fmaxf(red[tid], red[tid+s]); __syncthreads(); }
    m = red[0];
    __syncthreads();
    float sum = 0.f;
    for (int p = tid; p < NPAIR; p += 256) sum += expf(logits[p*32 + c] - m);
    red[tid] = sum; __syncthreads();
    for (int s = 128; s > 0; s >>= 1) { if (tid < s) red[tid] += red[tid+s]; __syncthreads(); }
    const float ls = m + logf(red[0]);
    for (int p = tid; p < NPAIR; p += 256) out[p*32 + c] = logits[p*32 + c] - ls;
}

// ---------------------------------------------------------------------------
extern "C" void kernel_launch(void* const* d_in, const int* in_sizes, int n_in,
                              void* d_out, int out_size, void* d_ws, size_t ws_size,
                              hipStream_t stream) {
    const float* x   = (const float*)d_in[0];
    const float* h0  = (const float*)d_in[1];
    const float* Wih = (const float*)d_in[2];
    const float* Whh = (const float*)d_in[3];
    const float* bih = (const float*)d_in[4];
    const float* bhh = (const float*)d_in[5];
    const float* W1  = (const float*)d_in[6];
    const float* b1  = (const float*)d_in[7];
    const float* W2  = (const float*)d_in[8];
    const float* b2  = (const float*)d_in[9];
    const float* W3  = (const float*)d_in[10];
    const float* b3  = (const float*)d_in[11];
    const float* Wt  = (const float*)d_in[12];
    const float* bt  = (const float*)d_in[13];
    float* out = (float*)d_out;

    // Workspace layout (~7.6 MB peak; everything L2-resident).
    // gia (2.4 MB) aliases the logits region: gia is dead after gru_kernel,
    // logits is first written by mlp_kernel (strictly later on the stream).
    char* ws = (char*)d_ws;
    float*    gout   = (float*)(ws);                                   // 0.82 MB GRU out (S,B,H)
    float*    P      = (float*)(ws + (size_t)1*1024*1024);             // 2 MB
    float*    Q      = (float*)(ws + (size_t)3*1024*1024);             // 2 MB
    _Float16* w2t    = (_Float16*)(ws + (size_t)5*1024*1024);          // 128 KB
    _Float16* w3h    = (_Float16*)(ws + (size_t)5*1024*1024 + 128*1024);   // 8 KB
    float*    logits = (float*)(ws + (size_t)5*1024*1024 + 256*1024);  // 2 MB
    float*    gia    = (float*)(ws + (size_t)5*1024*1024 + 256*1024);  // 2.4 MB (aliases logits)

    gia_kernel<<<SEQ*BAT, 320, 0, stream>>>(x, Wih, bih, gia);
    gru_kernel<<<BAT, 320, 0, stream>>>(gia, h0, Whh, bhh, gout);
    pq_kernel<<<SEQ*BAT, 256, 0, stream>>>(gout, W1, P, Q);
    w2t_kernel<<<256, 256, 0, stream>>>(W2, w2t);
    w3h_kernel<<<16, 256, 0, stream>>>(W3, w3h);
    mlp_kernel<<<NROWS/128, 256, 0, stream>>>(P, Q, w2t, w3h, b1, b2, b3, Wt, bt, logits);
    lsm_kernel<<<BAT*NOUT, 256, 0, stream>>>(logits, out);
}
